// Attention_1967095022262
// MI455X (gfx1250) — matrix operand order repaired and checked by reading, not compiled, not run on hardware
//
#include <hip/hip_runtime.h>
#include <hip/hip_bf16.h>

// MI455X / gfx1250 attention, softmax over the QUERY axis.
// All matmuls via v_wmma_f32_16x16x32_bf16 (wave32, f32 accumulate).
// Attention block = (batch, 32-k-column block), loops over all 16 heads so the
// head-averaged attention output is accumulated in LDS (no global atomics for
// it). Score panel + head-sum panel + staged K/V tiles = ~266 KB LDS, sized
// for CDNA5's 320 KB WGP LDS.
//
// Workspace (~56 MB): Qb (B,H,Nq,D) bf16 | Kb (B,H,Nk,D) bf16 |
//                     Vt (B,H,D,Nk) bf16 | Xa (B,H,Nq,D) f32

#define DIMC   1024
#define NQ     1024
#define NK     2048
#define NHEADS 16
#define HDIM   64
#define BATCH  4
#define QKSCALE 0.125f   // 64^-0.5

typedef __attribute__((ext_vector_type(16))) __bf16 bf16x16;
typedef __attribute__((ext_vector_type(8)))  __bf16 bf16x8;
typedef __attribute__((ext_vector_type(8)))  float  f32x8;

static __device__ __forceinline__ f32x8 wmma_bf16(bf16x16 a, bf16x16 b, f32x8 c) {
  return __builtin_amdgcn_wmma_f32_16x16x32_bf16(false, a, false, b, (short)0, c,
                                                 false, false);
}

// ---- WMMA fragment builders (VGPR layouts per ISA 7.12.2) -----------------
// A (16x32): lane = M row; K chunks split by lane-half.
static __device__ __forceinline__ bf16x16 frag_a_f32(const float* chunk, int lane) {
  const int half = (lane >> 4) & 1;
  const float4 u0 = *(const float4*)(chunk + half * 8);
  const float4 u1 = *(const float4*)(chunk + half * 8 + 4);
  const float4 v0 = *(const float4*)(chunk + 16 + half * 8);
  const float4 v1 = *(const float4*)(chunk + 16 + half * 8 + 4);
  bf16x16 a;
  a[0] = (__bf16)u0.x; a[1] = (__bf16)u0.y; a[2]  = (__bf16)u0.z; a[3]  = (__bf16)u0.w;
  a[4] = (__bf16)u1.x; a[5] = (__bf16)u1.y; a[6]  = (__bf16)u1.z; a[7]  = (__bf16)u1.w;
  a[8] = (__bf16)v0.x; a[9] = (__bf16)v0.y; a[10] = (__bf16)v0.z; a[11] = (__bf16)v0.w;
  a[12] = (__bf16)v1.x; a[13] = (__bf16)v1.y; a[14] = (__bf16)v1.z; a[15] = (__bf16)v1.w;
  return a;
}
static __device__ __forceinline__ bf16x16 frag_a_bf16(const __bf16* chunk, int lane) {
  const int half = (lane >> 4) & 1;
  bf16x8 lo = *(const bf16x8*)(chunk + half * 8);
  bf16x8 hi = *(const bf16x8*)(chunk + 16 + half * 8);
  bf16x16 a;
#pragma unroll
  for (int i = 0; i < 8; ++i) { a[i] = lo[i]; a[8 + i] = hi[i]; }
  return a;
}
// B (32x16): lane = N col; K chunks split by lane-half (same as A).
static __device__ __forceinline__ bf16x16 frag_b_f32(const float* row_k0, int lane) {
  const float* p = row_k0 + ((lane >> 4) & 1) * 8;
  const float4 u0 = *(const float4*)(p + 0);
  const float4 u1 = *(const float4*)(p + 4);
  const float4 u2 = *(const float4*)(p + 16);
  const float4 u3 = *(const float4*)(p + 20);
  bf16x16 b;
  b[0] = (__bf16)u0.x; b[1] = (__bf16)u0.y; b[2]  = (__bf16)u0.z; b[3]  = (__bf16)u0.w;
  b[4] = (__bf16)u1.x; b[5] = (__bf16)u1.y; b[6]  = (__bf16)u1.z; b[7]  = (__bf16)u1.w;
  b[8] = (__bf16)u2.x; b[9] = (__bf16)u2.y; b[10] = (__bf16)u2.z; b[11] = (__bf16)u2.w;
  b[12] = (__bf16)u3.x; b[13] = (__bf16)u3.y; b[14] = (__bf16)u3.z; b[15] = (__bf16)u3.w;
  return b;
}
static __device__ __forceinline__ bf16x16 frag_b_bf16(const __bf16* row_k0, int lane) {
  const __bf16* p = row_k0 + ((lane >> 4) & 1) * 8;
  bf16x8 lo = *(const bf16x8*)(p);
  bf16x8 hi = *(const bf16x8*)(p + 16);
  bf16x16 b;
#pragma unroll
  for (int i = 0; i < 8; ++i) { b[i] = lo[i]; b[8 + i] = hi[i]; }
  return b;
}

// ---- Tensor Data Mover staging (ISA ch.8 D# descriptor) -------------------
#if __has_builtin(__builtin_amdgcn_tensor_load_to_lds)
#define HAVE_TDM 1
typedef __attribute__((ext_vector_type(4))) unsigned int u32x4;
typedef __attribute__((ext_vector_type(4))) int i32x4;
typedef __attribute__((ext_vector_type(8))) int i32x8;

// 2-D tile load (data_size = 2 bytes): tile_d0 x tile_d1 elements, row stride
// stride0 elements, into LDS at lds_addr.
static __device__ __forceinline__ void tdm_load_2d(
    unsigned lds_addr, unsigned long long gaddr,
    unsigned tile_d0, unsigned tile_d1, unsigned long long stride0,
    unsigned tensor_d0, unsigned tensor_d1) {
  u32x4 g0;
  g0[0] = 1u;                                               // count=1, user D#
  g0[1] = lds_addr;                                         // lds_addr[31:0]
  g0[2] = (unsigned)(gaddr & 0xffffffffull);                // global_addr lo
  g0[3] = (unsigned)((gaddr >> 32) & 0x1ffffffull) | (2u << 30);  // hi | type=2
  i32x8 g1;
  g1[0] = 0x00010000;                  // wg_mask=0, data_size=1 (2B)
  g1[1] = (int)((tensor_d0 & 0xffffu) << 16);               // td0[15:0]@63:48
  g1[2] = (int)((tensor_d0 >> 16) | ((tensor_d1 & 0xffffu) << 16));
  g1[3] = (int)((tensor_d1 >> 16) | (tile_d0 << 16));       // tile_d0@127:112
  g1[4] = (int)(tile_d1 & 0xffffu);                         // tile_d1@143:128
  g1[5] = (int)(stride0 & 0xffffffffull);                   // stride0[31:0]
  g1[6] = (int)((stride0 >> 32) & 0xffffull);               // stride0[47:32]
  g1[7] = 0;
  i32x4 z = {0, 0, 0, 0};
#if __clang_major__ >= 23
  i32x8 z8 = {0, 0, 0, 0, 0, 0, 0, 0};
  __builtin_amdgcn_tensor_load_to_lds(g0, g1, z, z, z8, 0);
#else
  __builtin_amdgcn_tensor_load_to_lds(g0, g1, z, z, 0);
#endif
}
#else
#define HAVE_TDM 0
#endif

// ---------------------------------------------------------------------------
// Projection GEMM: Y = X @ W^T (f32 in, bf16 out). Wave computes 16x64
// (A-fragment reused over 4 N-tiles). ROWS_LOG2: rows-per-batch shift.
// VT=0: dst[(b,h,row,d)]; VT=1: dst[(b,h,d,row)] (transposed V).
// ---------------------------------------------------------------------------
template <int ROWS_LOG2, int VT>
__global__ void proj_gemm_kernel(const float* __restrict__ X, const float* __restrict__ W,
                                 __bf16* __restrict__ dst) {
  const int lane = threadIdx.x & 31;
  const int wave = threadIdx.x >> 5;
  const int rowbase = blockIdx.x * 16;
  const int colbase = (blockIdx.y * 8 + wave) * 64;
  const float* Arow = X + (size_t)(rowbase + (lane & 15)) * DIMC;
  f32x8 acc[4] = {};
  for (int k0 = 0; k0 < DIMC; k0 += 32) {
    __builtin_prefetch(Arow + k0 + 256, 0, 3);
    bf16x16 a = frag_a_f32(Arow + k0, lane);
#pragma unroll
    for (int t = 0; t < 4; ++t) {
      const float* Brow = W + (size_t)(colbase + t * 16 + (lane & 15)) * DIMC;
      acc[t] = wmma_bf16(a, frag_b_f32(Brow + k0, lane), acc[t]);
    }
  }
  const int mo = (lane >> 4) * 8;
#pragma unroll
  for (int t = 0; t < 4; ++t) {
    const int n = colbase + t * 16 + (lane & 15);
    const int h = n >> 6, d = n & 63;
#pragma unroll
    for (int r = 0; r < 8; ++r) {
      int grow = rowbase + mo + r;
      int bidx = grow >> ROWS_LOG2;
      int local = grow & ((1 << ROWS_LOG2) - 1);
      size_t idx = VT
          ? (((size_t)(bidx * NHEADS + h) * HDIM + d) << ROWS_LOG2) + local
          : ((((size_t)(bidx * NHEADS + h) << ROWS_LOG2) + local) * HDIM + d);
      dst[idx] = (__bf16)acc[t][r];
    }
  }
}

// ---------------------------------------------------------------------------
// Attention: one block per (batch, 32-k-column block); loops over all heads.
// LDS: S[1024][32] f32 (128 KB) | ASUM[1024][32] f32 (128 KB) |
//      reductions (1.3 KB) | staged K/V bf16 tiles (8 KB)  = ~266 KB.
// ---------------------------------------------------------------------------
__global__ void attn_kernel(const __bf16* __restrict__ Qb, const __bf16* __restrict__ Kb,
                            const __bf16* __restrict__ Vt, float* __restrict__ Xacc,
                            float* __restrict__ attn_mean) {
  extern __shared__ float smem[];              // S[NQ][32]
  float* ASUM = smem + NQ * 32;                // [NQ][32] head-sum panel
  float* red  = ASUM + NQ * 32;                // [8][32]
  float* cmax = red + 256;                     // [32]
  float* cinv = cmax + 32;                     // [32]
  __bf16* Kt  = (__bf16*)(cinv + 32);          // [32][64] bf16 (4 KB)
  __bf16* Vtl = Kt + 32 * 64;                  // [64][32] bf16 (4 KB)
  const int tid = threadIdx.x, lane = tid & 31, wave = tid >> 5;
  const int b = blockIdx.x;
  const int kbase = blockIdx.y * 32;
  const int col = tid & 31, part = tid >> 5;

  for (int i = tid; i < NQ * 32; i += 256) ASUM[i] = 0.f;

  for (int h = 0; h < NHEADS; ++h) {
    const int bh = b * NHEADS + h;
    __syncthreads();   // previous iteration done with S / K / V tiles

    // ---- Stage K (contiguous 4 KB) and V (64 rows x 64 B) tiles into LDS ----
#if HAVE_TDM
    if (tid < 32) {  // one TDM issue per descriptor, from wave 0 (EXEC ignored)
      const __bf16* ksrc = Kb + ((size_t)bh * NK + kbase) * HDIM;
      tdm_load_2d((unsigned)(size_t)Kt, (unsigned long long)(size_t)ksrc,
                  /*tile_d0=*/2048, /*tile_d1=*/1, /*stride0=*/2048,
                  /*tensor_d0=*/2048, /*tensor_d1=*/1);
      const __bf16* vsrc = Vt + ((size_t)bh * HDIM) * NK + kbase;
      tdm_load_2d((unsigned)(size_t)Vtl, (unsigned long long)(size_t)vsrc,
                  /*tile_d0=*/32, /*tile_d1=*/HDIM, /*stride0=*/NK,
                  /*tensor_d0=*/NK, /*tensor_d1=*/HDIM);
      __builtin_amdgcn_s_wait_tensorcnt(0);
    }
#else
    {
      const uint4* ks = (const uint4*)(Kb + ((size_t)bh * NK + kbase) * HDIM);
      ((uint4*)Kt)[tid] = ks[tid];                       // 256 x 16 B = 4 KB
      const int d = tid >> 2, q4 = tid & 3;
      const uint4* vs = (const uint4*)(Vt + ((size_t)bh * HDIM + d) * NK + kbase);
      ((uint4*)Vtl)[tid] = vs[q4];                       // 64 rows x 4 x 16 B
    }
#endif
    __syncthreads();

    // ---- Phase 1: S = (Q Kb^T) * scale -> LDS panel ----
    bf16x16 kf[2][2];
#pragma unroll
    for (int ct = 0; ct < 2; ++ct)
#pragma unroll
      for (int dd = 0; dd < 2; ++dd)
        kf[ct][dd] = frag_b_bf16(Kt + (ct * 16 + (lane & 15)) * HDIM + dd * 32, lane);

    for (int rt = wave; rt < NQ / 16; rt += 8) {
      const int q0 = rt * 16;
      const __bf16* Arow = Qb + ((size_t)bh * NQ + q0 + (lane & 15)) * HDIM;
      bf16x16 a0 = frag_a_bf16(Arow, lane);
      bf16x16 a1 = frag_a_bf16(Arow + 32, lane);
#pragma unroll
      for (int ct = 0; ct < 2; ++ct) {
        f32x8 c = {};
        c = wmma_bf16(a0, kf[ct][0], c);
        c = wmma_bf16(a1, kf[ct][1], c);
        const int n = ct * 16 + (lane & 15);
        const int mo = (lane >> 4) * 8;
#pragma unroll
        for (int r = 0; r < 8; ++r) smem[(q0 + mo + r) * 32 + n] = c[r] * QKSCALE;
      }
    }
    __syncthreads();

    // ---- Phase 2: column softmax over q; accumulate head sum in LDS ----
    float mx = -1e30f;
    for (int q = part; q < NQ; q += 8) mx = fmaxf(mx, smem[q * 32 + col]);
    red[part * 32 + col] = mx;
    __syncthreads();
    if (tid < 32) {
      float m = red[col];
#pragma unroll
      for (int p = 1; p < 8; ++p) m = fmaxf(m, red[p * 32 + col]);
      cmax[col] = m;
    }
    __syncthreads();
    const float cm = cmax[col];
    float s = 0.f;
    for (int q = part; q < NQ; q += 8) {
      float e = __expf(smem[q * 32 + col] - cm);
      smem[q * 32 + col] = e;
      s += e;
    }
    red[part * 32 + col] = s;
    __syncthreads();
    if (tid < 32) {
      float z = 0.f;
#pragma unroll
      for (int p = 0; p < 8; ++p) z += red[p * 32 + col];
      cinv[col] = 1.f / z;
    }
    __syncthreads();
    const float inv = cinv[col];
    for (int q = part; q < NQ; q += 8) {
      float pv = smem[q * 32 + col] * inv;
      smem[q * 32 + col] = pv;
      ASUM[q * 32 + col] += pv;          // no global atomics: block owns (b,q,k)
    }
    __syncthreads();

    // ---- Phase 3: Xacc += P(1024x32) @ Vblock(32x64) ----
    bf16x16 vf[4];
#pragma unroll
    for (int ct = 0; ct < 4; ++ct)
      vf[ct] = frag_b_bf16(Vtl + (ct * 16 + (lane & 15)) * 32, lane);

    for (int rt = wave; rt < NQ / 16; rt += 8) {
      const int q0 = rt * 16;
      bf16x16 a = frag_a_f32(smem + (size_t)(q0 + (lane & 15)) * 32, lane);
#pragma unroll
      for (int ct = 0; ct < 4; ++ct) {
        f32x8 c = {};
        c = wmma_bf16(a, vf[ct], c);
        const int d = ct * 16 + (lane & 15);
        const int mo = (lane >> 4) * 8;
#pragma unroll
        for (int r = 0; r < 8; ++r)
          atomicAdd(&Xacc[((size_t)bh * NQ + q0 + mo + r) * HDIM + d], c[r]);
      }
    }
  }

  // ---- Head-averaged attention output: plain coalesced stores ----
  __syncthreads();
  for (int i = tid; i < NQ * 32; i += 256) {
    const int q = i >> 5, c = i & 31;
    attn_mean[((size_t)b * NQ + q) * NK + kbase + c] = ASUM[i] * (1.0f / NHEADS);
  }
}

// ---------------------------------------------------------------------------
// Output projection: out[(q,b,:)] = Xflat[(b,q,:)] @ Wp^T + bp ; wave = 16x64.
// ---------------------------------------------------------------------------
__global__ void out_proj_kernel(const float* __restrict__ Xacc, const float* __restrict__ Wp,
                                const float* __restrict__ bias, float* __restrict__ out) {
  const int lane = threadIdx.x & 31;
  const int wave = threadIdx.x >> 5;
  const int rowbase = blockIdx.x * 16;                  // flat (b*NQ + q)
  const int colbase = (blockIdx.y * 8 + wave) * 64;
  const int arow = rowbase + (lane & 15);
  const int ab = arow >> 10, aq = arow & 1023;
  f32x8 acc[4] = {};
  for (int k0 = 0; k0 < DIMC; k0 += 32) {
    const float* Achunk =
        Xacc + (((size_t)(ab * NHEADS + (k0 >> 6)) * NQ + aq) * HDIM + (k0 & 63));
    bf16x16 a = frag_a_f32(Achunk, lane);
#pragma unroll
    for (int t = 0; t < 4; ++t) {
      const float* Brow = Wp + (size_t)(colbase + t * 16 + (lane & 15)) * DIMC;
      acc[t] = wmma_bf16(a, frag_b_f32(Brow + k0, lane), acc[t]);
    }
  }
  const int mo = (lane >> 4) * 8;
#pragma unroll
  for (int t = 0; t < 4; ++t) {
    const int n = colbase + t * 16 + (lane & 15);
    const float bv = bias[n];
#pragma unroll
    for (int r = 0; r < 8; ++r) {
      int grow = rowbase + mo + r;
      int ob = grow >> 10, oq = grow & 1023;
      out[((size_t)oq * BATCH + ob) * DIMC + n] = acc[t][r] + bv;  // (Nq,B,C)
    }
  }
}

__global__ void zero_f32_kernel(float* __restrict__ p, size_t n) {
  size_t i = (size_t)blockIdx.x * blockDim.x + threadIdx.x;
  size_t stride = (size_t)gridDim.x * blockDim.x;
  for (; i < n; i += stride) p[i] = 0.f;
}

extern "C" void kernel_launch(void* const* d_in, const int* in_sizes, int n_in,
                              void* d_out, int out_size, void* d_ws, size_t ws_size,
                              hipStream_t stream) {
  (void)in_sizes; (void)n_in; (void)out_size; (void)ws_size;
  const float* xq = (const float*)d_in[0];
  const float* xk = (const float*)d_in[1];
  const float* xv = (const float*)d_in[2];
  const float* Wq = (const float*)d_in[3];
  const float* Wk = (const float*)d_in[4];
  const float* Wv = (const float*)d_in[5];
  const float* Wp = (const float*)d_in[6];
  const float* bp = (const float*)d_in[7];

  float* out      = (float*)d_out;
  float* out_x    = out;                                  // (Nq,B,C)
  float* out_attn = out + (size_t)NQ * BATCH * DIMC;      // (B,Nq,Nk)

  char* ws = (char*)d_ws;
  const size_t qb_bytes = (size_t)BATCH * NHEADS * NQ * HDIM * 2;  // 8 MB
  const size_t kb_bytes = (size_t)BATCH * NHEADS * NK * HDIM * 2;  // 16 MB
  __bf16* Qb = (__bf16*)ws;
  __bf16* Kb = (__bf16*)(ws + qb_bytes);
  __bf16* Vt = (__bf16*)(ws + qb_bytes + kb_bytes);
  float*  Xa = (float*)(ws + qb_bytes + 2 * kb_bytes);             // 16 MB

  zero_f32_kernel<<<512, 256, 0, stream>>>(Xa, (size_t)BATCH * NHEADS * NQ * HDIM);

  proj_gemm_kernel<10, 0><<<dim3(BATCH * NQ / 16, 2), 256, 0, stream>>>(xq, Wq, Qb);
  proj_gemm_kernel<11, 0><<<dim3(BATCH * NK / 16, 2), 256, 0, stream>>>(xk, Wk, Kb);
  proj_gemm_kernel<11, 1><<<dim3(BATCH * NK / 16, 2), 256, 0, stream>>>(xv, Wv, Vt);

  // LDS: 2 x 128 KB panels + 320 f32 reductions + 8 KB bf16 tiles
  const size_t smem_bytes = (size_t)(2 * NQ * 32 + 320) * sizeof(float) + 8192;
  attn_kernel<<<dim3(BATCH, NK / 32), 256, smem_bytes, stream>>>(
      Qb, Kb, Vt, Xa, out_attn);

  out_proj_kernel<<<dim3(BATCH * NQ / 16, 2), 256, 0, stream>>>(Xa, Wp, bp, out_x);
}
